// SpectralConv2d_1881195676181
// MI455X (gfx1250) — matrix operand order repaired and checked by reading, not compiled, not run on hardware
//
#include <hip/hip_runtime.h>
#include <hip/hip_bf16.h>
#include <math.h>

// ---------------- WMMA types / fragment helpers (gfx1250, wave32) ----------
typedef __attribute__((ext_vector_type(16))) _Float16 v16h;
typedef __attribute__((ext_vector_type(8)))  _Float16 v8h;
typedef __attribute__((ext_vector_type(4)))  _Float16 v4h;
typedef __attribute__((ext_vector_type(8)))  float    v8f;

// A fragment: 16x32 f16 tile, row-major in LDS, row stride ld (halfs, mult of 8).
// ISA 7.12.2: lane half 0 -> K {0..7,16..23}, half 1 -> K {8..15,24..31}:
// two contiguous 8-half runs per lane -> two ds_load_b128.
__device__ __forceinline__ v16h load_frag_a(const _Float16* lds, int ld) {
  int lane = threadIdx.x & 31;
  const _Float16* p = lds + (lane & 15) * ld + ((lane >> 4) << 3);
  v8h lo = *(const v8h*)(p);
  v8h hi = *(const v8h*)(p + 16);
  return __builtin_shufflevector(lo, hi, 0, 1, 2, 3, 4, 5, 6, 7,
                                 8, 9, 10, 11, 12, 13, 14, 15);
}

// B fragment: 32x16 (KxN) tile stored N-MAJOR in LDS: lds[n][k], row stride
// ldk (halfs, mult of 8). Lane = column, half 0 -> K {0..7,16..23}, half 1 -> K {8..15,24..31}:
// two contiguous 8-half runs per lane -> two ds_load_b128.
__device__ __forceinline__ v16h load_frag_b(const _Float16* lds, int ldk) {
  int lane = threadIdx.x & 31;
  const _Float16* p = lds + (lane & 15) * ldk + ((lane >> 4) << 3);
  v8h lo = *(const v8h*)(p);
  v8h hi = *(const v8h*)(p + 16);
  return __builtin_shufflevector(lo, hi, 0, 1, 2, 3, 4, 5, 6, 7,
                                 8, 9, 10, 11, 12, 13, 14, 15);
}

#define WMMA_F16(a, b, c) \
  __builtin_amdgcn_wmma_f32_16x16x32_f16(false, (a), false, (b), (short)0, (c), false, false)

// ---------------- Problem constants -----------------
#define MX 21
#define MT 21
// workspace offsets in _Float16 units
constexpr size_t OFF_PT  = 0;                       // 64*128   : DFT-W, N-major [j2][m], incl 1/128
constexpr size_t OFF_PX2 = OFF_PT  + 64 * 128;      // 48*256   : DFT-H complex-encoded A [i2][n2]
constexpr size_t OFF_QX  = OFF_PX2 + 48 * 256;      // 128*64   : iDFT-H A [n][i2]
constexpr size_t OFF_QT  = OFF_QX  + 128 * 64;      // 128*64   : iDFT-W, N-major [m][j2], incl 1/128 & Re() signs
constexpr size_t OFF_B3  = OFF_QT  + 128 * 64;      // 441*128*128 : weights, N-major [mode][o2][c2]
constexpr size_t OFF_T1  = OFF_B3  + 441 * 16384;   // 2048*8192 : stage1 out [bc][j][n2] (reused as T3)
constexpr size_t OFF_X   = OFF_T1  + 2048 * 8192;   // 2048*1536 : modes per (b,c) [i2][j]
constexpr size_t OFF_OK  = OFF_X   + 2048 * 1536;   // 2048*2048 : mixed modes per (b,o) [i][j2]
// total = 31,379,456 halfs = ~62.8 MB

// ---------------- Setup: phase matrices (tiny) -----------------
__global__ void k_setup(const float* __restrict__ thx_raw,
                        const float* __restrict__ tht_raw,
                        _Float16* __restrict__ ws) {
  const float PI = 3.14159265358979f;
  _Float16* PT  = ws + OFF_PT;
  _Float16* PX2 = ws + OFF_PX2;
  _Float16* QX  = ws + OFF_QX;
  _Float16* QT  = ws + OFF_QT;
  int tid = threadIdx.x;

  // PT (N-major) [j2][m] : col 2j = cos, 2j+1 = sin of -2*pi*k_j*m/128, * (1/128)
  for (int t = tid; t < 64 * 128; t += blockDim.x) {
    int j2 = t >> 7, m = t & 127, j = j2 >> 1;
    float v = 0.f;
    if (j < MT) {
      float k = (j <= 10) ? (float)j : (float)(j - 21);
      float ang = -2.f * PI * k * (float)m * (1.f / 128.f);
      v = ((j2 & 1) ? sinf(ang) : cosf(ang)) * (1.f / 128.f);
    }
    PT[t] = (_Float16)v;
  }
  // PX2 (A, row-major) [i2][n2] : row 2i = (pxr,-pxi), row 2i+1 = (pxi,pxr)
  for (int t = tid; t < 48 * 256; t += blockDim.x) {
    int i2 = t >> 8, n2 = t & 255;
    int i = i2 >> 1, si = i2 & 1, n = n2 >> 1, sn = n2 & 1;
    float v = 0.f;
    if (i < MX) {
      float k = (i <= 10) ? (float)i : (float)(i - 21);
      float ang = -2.f * PI * k * (float)n * (1.f / 128.f);
      float c = cosf(ang), s = sinf(ang);
      v = (si == 0) ? ((sn == 0) ? c : -s) : ((sn == 0) ? s : c);
    }
    PX2[t] = (_Float16)v;
  }
  // QX (A, row-major) [n][i2] : qx = e^{-w sin th}(cos(w cos th) + i sin(...)), w = 2 pi n|k|/128
  for (int t = tid; t < 128 * 64; t += blockDim.x) {
    int n = t >> 6, i2 = t & 63, i = i2 >> 1, r = i2 & 1;
    float v = 0.f;
    if (i < MX) {
      float k = (i <= 10) ? (float)i : (float)(i - 21);
      float sig = 1.f / (1.f + expf(-thx_raw[i]));
      float th = (k < 0.f) ? (PI * 0.5f) * (1.f + sig) : PI * sig;
      float w = 2.f * PI * (float)n * fabsf(k) * (1.f / 128.f);
      float e = expf(-w * sinf(th));
      float p = w * cosf(th);
      v = e * (r ? sinf(p) : cosf(p));
    }
    QX[t] = (_Float16)v;
  }
  // QT (N-major) [m][j2] : K row 2j = qtr, 2j+1 = -qti; * (1/128)
  for (int t = tid; t < 128 * 64; t += blockDim.x) {
    int m = t >> 6, j2 = t & 63, j = j2 >> 1, s = j2 & 1;
    float v = 0.f;
    if (j < MT) {
      float k = (j <= 10) ? (float)j : (float)(j - 21);
      float sig = 1.f / (1.f + expf(-tht_raw[j]));
      float th = (k < 0.f) ? (PI * 0.5f) * (1.f + sig) : PI * sig;
      float w = 2.f * PI * (float)m * fabsf(k) * (1.f / 128.f);
      float e = expf(-w * sinf(th)) * (1.f / 128.f);
      float p = w * cosf(th);
      v = s ? (-e * sinf(p)) : (e * cosf(p));
    }
    QT[t] = (_Float16)v;
  }
}

// ---------------- Repack weights, N-major K-doubled form -------------------
// B3[mode][o2][c2] : K rows c2 = (Rre Rim ; -Rim Rre) structure
__global__ void k_repack(const float* __restrict__ wr, const float* __restrict__ wi,
                         _Float16* __restrict__ B3) {
  int idx = blockIdx.x * 256 + threadIdx.x;
  if (idx >= 441 * 16384) return;
  int mode = idx >> 14, rem = idx & 16383;
  int o2 = rem >> 7, c2 = rem & 127;
  int c = c2 >> 1, s = c2 & 1, o = o2 >> 1, r = o2 & 1;
  int i = mode / 21, j = mode % 21;
  int widx = ((c * 64 + o) * 21 + i) * 21 + j;  // (C,O,Mx,Mt)
  float re = wr[widx], im = wi[widx];
  float v = (s == 0) ? ((r == 0) ? re : im) : ((r == 0) ? -im : re);
  B3[idx] = (_Float16)v;
}

// ---------------- Stage 1: contract W. x(128x128) @ PT(128x42) per (b,c) ---
__global__ __launch_bounds__(256) void k_dftw(const float* __restrict__ x,
                                              const _Float16* __restrict__ wsPT,
                                              _Float16* __restrict__ T1) {
  __shared__ __align__(16) _Float16 sX[128 * 128];  // A, row-major, 32 KB
  __shared__ __align__(16) _Float16 sPT[64 * 128];  // B, N-major [j2][m], 16 KB
  int bc = blockIdx.x, tid = threadIdx.x;
  const float* xp = x + (size_t)bc * 16384;
  for (int t = tid; t < 4096; t += 256) {
    float4 f = ((const float4*)xp)[t];
    v4h h = {(_Float16)f.x, (_Float16)f.y, (_Float16)f.z, (_Float16)f.w};
    *(v4h*)(sX + t * 4) = h;
  }
  for (int t = tid; t < 1024; t += 256) ((uint4*)sPT)[t] = ((const uint4*)wsPT)[t];
  __syncthreads();
  int wave = tid >> 5, lane = tid & 31, half = lane >> 4;
  v16h afr[4];
#pragma unroll
  for (int kk = 0; kk < 4; ++kk)
    afr[kk] = load_frag_a(sX + wave * 16 * 128 + kk * 32, 128);
  _Float16* T1p = T1 + (size_t)bc * 8192;
  for (int nt = 0; nt < 4; ++nt) {
    v8f acc = {};
#pragma unroll
    for (int kk = 0; kk < 4; ++kk)
      acc = WMMA_F16(afr[kk], load_frag_b(sPT + nt * 16 * 128 + kk * 32, 128), acc);
    int j2 = nt * 16 + (lane & 15);
    int j = j2 >> 1, s = j2 & 1;
    if (j < MT) {
#pragma unroll
      for (int v = 0; v < 8; ++v) {
        int n = wave * 16 + half * 8 + v;
        T1p[j * 256 + 2 * n + s] = (_Float16)acc[v];  // [j][n2] for stage-2 B fill
      }
    }
  }
}

// ---------------- Stage 2: contract H (complex). PX2(42x256) @ T1bc(256x21) -
__global__ __launch_bounds__(192) void k_dfth(const _Float16* __restrict__ wsPX2,
                                              const _Float16* __restrict__ T1,
                                              _Float16* __restrict__ X) {
  __shared__ __align__(16) _Float16 sPX[48 * 256];  // A, row-major, 24 KB
  __shared__ __align__(16) _Float16 sT[32 * 256];   // B, N-major [j][n2], 16 KB
  int bc = blockIdx.x, tid = threadIdx.x;
  for (int t = tid; t < 1536; t += 192) ((uint4*)sPX)[t] = ((const uint4*)wsPX2)[t];
  const _Float16* T1p = T1 + (size_t)bc * 8192;
  for (int t = tid; t < 1024; t += 192) ((uint4*)sT)[t] = ((const uint4*)T1p)[t];
  __syncthreads();
  int wave = tid >> 5, lane = tid & 31, half = lane >> 4;
  int mt = wave >> 1, nt = wave & 1;  // 3 M-tiles x 2 N-tiles over 6 waves
  v8f acc = {};
#pragma unroll
  for (int kk = 0; kk < 8; ++kk)
    acc = WMMA_F16(load_frag_a(sPX + mt * 16 * 256 + kk * 32, 256),
                   load_frag_b(sT + nt * 16 * 256 + kk * 32, 256), acc);
  _Float16* Xp = X + (size_t)bc * 1536;  // [i2(48)][j(32)]
  int j = nt * 16 + (lane & 15);
#pragma unroll
  for (int v = 0; v < 8; ++v) {
    int i2 = mt * 16 + half * 8 + v;
    Xp[i2 * 32 + j] = (_Float16)acc[v];
  }
}

// ---------------- Stage 3: per-mode complex channel mix (441 GEMMs) --------
__global__ __launch_bounds__(256) void k_mix(const _Float16* __restrict__ X,
                                             const _Float16* __restrict__ B3,
                                             _Float16* __restrict__ OK) {
  __shared__ __align__(16) _Float16 sA[32 * 128];   // A[b][c2], 8 KB
  __shared__ __align__(16) _Float16 sB[128 * 128];  // B, N-major [o2][c2], 32 KB
  int mode = blockIdx.x, tid = threadIdx.x;
  int i = mode / 21, j = mode % 21;
  for (int t = tid; t < 4096; t += 256) {
    int b = t >> 7, c2 = t & 127;
    sA[t] = X[(size_t)((b << 6) + (c2 >> 1)) * 1536 + (2 * i + (c2 & 1)) * 32 + j];
  }
  const _Float16* Bp = B3 + (size_t)mode * 16384;
  for (int t = tid; t < 2048; t += 256) ((uint4*)sB)[t] = ((const uint4*)Bp)[t];
  __syncthreads();
  int wave = tid >> 5, lane = tid & 31, half = lane >> 4;
  int mt = (wave * 2) >> 3;  // same M-tile for both of this wave's tiles
  v16h afr[4];
#pragma unroll
  for (int kk = 0; kk < 4; ++kk)
    afr[kk] = load_frag_a(sA + mt * 16 * 128 + kk * 32, 128);
#pragma unroll
  for (int q = 0; q < 2; ++q) {
    int nt = (wave * 2 + q) & 7;
    v8f acc = {};
#pragma unroll
    for (int kk = 0; kk < 4; ++kk)
      acc = WMMA_F16(afr[kk], load_frag_b(sB + nt * 16 * 128 + kk * 32, 128), acc);
    int o2 = nt * 16 + (lane & 15);
    int o = o2 >> 1, r = o2 & 1;
#pragma unroll
    for (int v = 0; v < 8; ++v) {
      int b = mt * 16 + half * 8 + v;
      OK[(size_t)((b << 6) + o) * 2048 + i * 64 + 2 * j + r] = (_Float16)acc[v];
    }
  }
}

// ---------------- Stage 4: inverse over H per (b,o). QX(128x42)@OK(42x42) ---
__global__ __launch_bounds__(256) void k_idfth(const _Float16* __restrict__ wsQX,
                                               const _Float16* __restrict__ OK,
                                               _Float16* __restrict__ T3) {
  __shared__ __align__(16) _Float16 sQX[128 * 64];  // A, row-major, 16 KB
  __shared__ __align__(16) _Float16 sB4[64 * 64];   // B, N-major [j2][i2], 8 KB
  int bo = blockIdx.x, tid = threadIdx.x;
  for (int t = tid; t < 1024; t += 256) ((uint4*)sQX)[t] = ((const uint4*)wsQX)[t];
  const _Float16* OKp = OK + (size_t)bo * 2048;
  for (int t = tid; t < 4096; t += 256) {  // K-doubled complex B, built transposed
    int j2 = t >> 6, i2 = t & 63;
    int i = i2 >> 1, s = i2 & 1, j = j2 >> 1, r = j2 & 1;
    float v = 0.f;
    if (i < MX && j < MT) {
      float re = (float)OKp[i * 64 + 2 * j];
      float im = (float)OKp[i * 64 + 2 * j + 1];
      v = (s == 0) ? ((r == 0) ? re : im) : ((r == 0) ? -im : re);
    }
    sB4[t] = (_Float16)v;
  }
  __syncthreads();
  int wave = tid >> 5, lane = tid & 31, half = lane >> 4;
  _Float16* T3p = T3 + (size_t)bo * 8192;
  v16h afr[2];
#pragma unroll
  for (int kk = 0; kk < 2; ++kk)
    afr[kk] = load_frag_a(sQX + wave * 16 * 64 + kk * 32, 64);  // mt == wave
#pragma unroll
  for (int nt = 0; nt < 4; ++nt) {
    v8f acc = {};
#pragma unroll
    for (int kk = 0; kk < 2; ++kk)
      acc = WMMA_F16(afr[kk], load_frag_b(sB4 + nt * 16 * 64 + kk * 32, 64), acc);
    int j2 = nt * 16 + (lane & 15);
#pragma unroll
    for (int v = 0; v < 8; ++v) {
      int n = wave * 16 + half * 8 + v;
      T3p[n * 64 + j2] = (_Float16)acc[v];
    }
  }
}

// ---------------- Stage 5: inverse over W, real out. T3(128x42)@QT(42x128) --
__global__ __launch_bounds__(256) void k_idftw(const _Float16* __restrict__ T3,
                                               const _Float16* __restrict__ wsQT,
                                               float* __restrict__ out) {
  __shared__ __align__(16) _Float16 sA[128 * 64];   // A, row-major, 16 KB
  __shared__ __align__(16) _Float16 sQT[128 * 64];  // B, N-major [m][j2], 16 KB
  int slab = blockIdx.x, tid = threadIdx.x;
  const _Float16* Tp = T3 + (size_t)slab * 8192;
  for (int t = tid; t < 1024; t += 256) ((uint4*)sA)[t] = ((const uint4*)Tp)[t];
  for (int t = tid; t < 1024; t += 256) ((uint4*)sQT)[t] = ((const uint4*)wsQT)[t];
  __syncthreads();
  int wave = tid >> 5, lane = tid & 31, half = lane >> 4;
  float* op = out + (size_t)slab * 16384;
  v16h afr[2];
#pragma unroll
  for (int kk = 0; kk < 2; ++kk)
    afr[kk] = load_frag_a(sA + wave * 16 * 64 + kk * 32, 64);
  for (int nt = 0; nt < 8; ++nt) {
    v8f acc = {};
#pragma unroll
    for (int kk = 0; kk < 2; ++kk)
      acc = WMMA_F16(afr[kk], load_frag_b(sQT + nt * 16 * 64 + kk * 32, 64), acc);
    int m = nt * 16 + (lane & 15);
#pragma unroll
    for (int v = 0; v < 8; ++v) {
      int n = wave * 16 + half * 8 + v;
      op[(size_t)n * 128 + m] = acc[v];
    }
  }
}

// ---------------- Launch -----------------
extern "C" void kernel_launch(void* const* d_in, const int* in_sizes, int n_in,
                              void* d_out, int out_size, void* d_ws, size_t ws_size,
                              hipStream_t stream) {
  const float* x   = (const float*)d_in[0];
  const float* wr  = (const float*)d_in[1];
  const float* wi  = (const float*)d_in[2];
  const float* thx = (const float*)d_in[3];
  const float* tht = (const float*)d_in[4];
  _Float16* ws = (_Float16*)d_ws;

  k_setup<<<1, 256, 0, stream>>>(thx, tht, ws);
  k_repack<<<(441 * 16384) / 256, 256, 0, stream>>>(wr, wi, ws + OFF_B3);
  k_dftw<<<2048, 256, 0, stream>>>(x, ws + OFF_PT, ws + OFF_T1);
  k_dfth<<<2048, 192, 0, stream>>>(ws + OFF_PX2, ws + OFF_T1, ws + OFF_X);
  k_mix<<<441, 256, 0, stream>>>(ws + OFF_X, ws + OFF_B3, ws + OFF_OK);
  // T3 aliases T1 (T1 is dead after k_dfth)
  k_idfth<<<2048, 256, 0, stream>>>(ws + OFF_QX, ws + OFF_OK, ws + OFF_T1);
  k_idftw<<<2048, 256, 0, stream>>>(ws + OFF_T1, ws + OFF_QT, (float*)d_out);
}
